// ModuleFusedSDPA_7103875908082
// MI455X (gfx1250) — matrix operand order repaired and checked by reading, not compiled, not run on hardware
//
#include <hip/hip_runtime.h>
#include <hip/hip_bf16.h>

// ---------------------------------------------------------------------------
// FlashAttention-style fused SDPA forward for gfx1250 (MI455X).
//   B=1, H=16, Q=4096, KV=8192, D=128, prefix = KV - Q = 4096 (causal suffix)
//   f32 in/out; f16 WMMA (v_wmma_f32_16x16x32_f16) for QK^T and PV;
//   f32 accumulation + online softmax; KV processed in 64-wide blocks;
//   double-buffered LDS operand fetch to hide DS latency under WMMA issue.
// ---------------------------------------------------------------------------

typedef __attribute__((ext_vector_type(2)))  _Float16     h2;
typedef __attribute__((ext_vector_type(4)))  float        f4;
typedef __attribute__((ext_vector_type(4)))  _Float16     h4;
typedef __attribute__((ext_vector_type(4)))  unsigned int u4;
typedef __attribute__((ext_vector_type(16))) _Float16     v16h;
typedef __attribute__((ext_vector_type(8)))  float        v8f;

#define NEG_INF_F (-1e30f)

// packed f32x2 -> f16x2 convert (v_cvt_pk_rtz_f16_f32); builtin returns an
// __fp16 vector, bit_cast to the _Float16 vector type we use everywhere.
__device__ __forceinline__ h2 pkh2(float a, float b) {
    return __builtin_bit_cast(h2, __builtin_amdgcn_cvt_pkrtz(a, b));
}

// ds_swizzle XOR-butterfly within 16-lane half-groups (wave32):
// pattern = and_mask 0x1F | xor_mask << 10
#define SWZ(x, pat) __int_as_float(__builtin_amdgcn_ds_swizzle(__float_as_int(x), (pat)))

__device__ __forceinline__ float redmax16(float x) {
    x = fmaxf(x, SWZ(x, 0x041F));
    x = fmaxf(x, SWZ(x, 0x081F));
    x = fmaxf(x, SWZ(x, 0x101F));
    x = fmaxf(x, SWZ(x, 0x201F));
    return x;
}
__device__ __forceinline__ float redsum16(float x) {
    x += SWZ(x, 0x041F);
    x += SWZ(x, 0x081F);
    x += SWZ(x, 0x101F);
    x += SWZ(x, 0x201F);
    return x;
}

// pack 16 f32 (two 8-float runs) into a WMMA f16 operand via packed converts
__device__ __forceinline__ v16h cvt16pk(f4 a, f4 b, f4 c, f4 d) {
    struct R { h2 x0, x1, x2, x3, x4, x5, x6, x7; } r;
    r.x0 = pkh2(a.x, a.y);
    r.x1 = pkh2(a.z, a.w);
    r.x2 = pkh2(b.x, b.y);
    r.x3 = pkh2(b.z, b.w);
    r.x4 = pkh2(c.x, c.y);
    r.x5 = pkh2(c.z, c.w);
    r.x6 = pkh2(d.x, d.y);
    r.x7 = pkh2(d.z, d.w);
    return __builtin_bit_cast(v16h, r);
}

__device__ __forceinline__ h4 cvt4pk(f4 a) {
    struct R { h2 lo, hi; } r;
    r.lo = pkh2(a.x, a.y);
    r.hi = pkh2(a.z, a.w);
    return __builtin_bit_cast(h4, r);
}

// two contiguous 16B LDS runs -> 16 f16 WMMA operand
__device__ __forceinline__ v16h lds16(const _Float16* p0, const _Float16* p1) {
    struct R { u4 a, b; } r;
    r.a = *(const u4*)p0;
    r.b = *(const u4*)p1;
    return __builtin_bit_cast(v16h, r);
}

namespace cfg {
constexpr int H    = 16;
constexpr int Q    = 4096;
constexpr int KV   = 8192;
constexpr int D    = 128;
constexpr int PREF = KV - Q;   // 4096
constexpr int QT   = 128;      // q rows per workgroup (8 waves * 16)
constexpr int KB   = 64;       // kv block per iteration (4 x 16-wide tiles)
}  // namespace cfg

__global__ __launch_bounds__(256, 1)
void fsdpa_fwd_kernel(const float* __restrict__ q,
                      const float* __restrict__ k,
                      const float* __restrict__ v,
                      const float* __restrict__ mask,
                      float* __restrict__ out) {
    using namespace cfg;

    // LDS staging (bank-padded row strides; all 16B-aligned vector runs)
    __shared__ __align__(16) _Float16 ldsK[64][136];    // K tile, [kv][d]
    __shared__ __align__(16) _Float16 ldsVt[128][72];   // V tile transposed, [d][kv]
    __shared__ __align__(16) _Float16 ldsP[8][16][72];  // per-wave P transpose scratch

    const int h    = blockIdx.y;
    const int q0wg = blockIdx.x * QT;
    const int tid  = threadIdx.x;
    const int wave = tid >> 5;
    const int lane = tid & 31;
    const int lg   = lane & 15;   // lane within half-group
    const int hs   = lane >> 4;   // half-group select
    const int qw   = q0wg + wave * 16;  // this wave's q base row

    const float scale = 0.08838834764831845f;  // 1/sqrt(128)

    // ---- preload Q rows (16 x 128) into WMMA A operands: 4 x v16h ----
    // A 16x32 f16 layout: lane m = lg; element e -> d = g*32 + hs*16 + e
    v16h aq[4];
    {
        const float* qr = q + ((size_t)h * Q + (qw + lg)) * D;
#pragma unroll
        for (int g = 0; g < 4; ++g) {
            const int d0 = g * 32 + hs * 16;
            const int d1 = g * 32 + hs * 16 + 8;
            f4 f0 = *(const f4*)(qr + d0);
            f4 f1 = *(const f4*)(qr + d0 + 4);
            f4 f2 = *(const f4*)(qr + d1);
            f4 f3 = *(const f4*)(qr + d1 + 4);
            aq[g] = cvt16pk(f0, f1, f2, f3);
        }
    }

    // ---- accumulator state ----
    v8f o[8];  // O tile: 16 q-rows x 128 d, as 8 WMMA C tiles (d chunks of 16)
#pragma unroll
    for (int dc = 0; dc < 8; ++dc) { v8f z = {}; o[dc] = z; }
    float mrow[8], lrow[8];  // per-lane: rows (i + 8*hs), replicated across half-group
#pragma unroll
    for (int i = 0; i < 8; ++i) { mrow[i] = -3.0e38f; lrow[i] = 0.0f; }

    const int kv_end_wg   = PREF + q0wg + QT;  // multiple of 64, <= KV
    const int kv_end_wave = PREF + qw + 16;

    for (int kvb = 0; kvb < kv_end_wg; kvb += KB) {
        // ---- cooperative stage: K tile (row-major) + V tile (transposed), f32->f16 ----
#pragma unroll
        for (int p = 0; p < 8; ++p) {
            const int idx = p * 1024 + tid * 4;   // 64*128 = 8192 elements total
            const int r   = idx >> 7;
            const int d   = idx & 127;
            const size_t gbase = ((size_t)h * KV + kvb + r) * D + d;
            f4 kf = *(const f4*)(k + gbase);
            f4 vf = *(const f4*)(v + gbase);
            *(h4*)&ldsK[r][d] = cvt4pk(kf);
            h2 v01 = pkh2(vf.x, vf.y);
            h2 v23 = pkh2(vf.z, vf.w);
            ldsVt[d + 0][r] = v01.x;
            ldsVt[d + 1][r] = v01.y;
            ldsVt[d + 2][r] = v23.x;
            ldsVt[d + 3][r] = v23.y;
        }
        __syncthreads();

        if (kvb < kv_end_wave) {  // wave-uniform causal skip
            // ---- S = Q K^T : four 16x16 tiles (kv cols kvb + t*16 + lg) ----
            // Double-buffered B-tile fetch: load k-step g+1 while WMMAs of g run.
            v8f s[4];
#pragma unroll
            for (int t = 0; t < 4; ++t) { v8f z = {}; s[t] = z; }

            v16h kb0[4], kb1[4];
            {
                const int db = hs * 16;  // g = 0
                kb0[0] = lds16(&ldsK[lg][db],      &ldsK[lg][db + 8]);
                kb0[1] = lds16(&ldsK[16 + lg][db], &ldsK[16 + lg][db + 8]);
                kb0[2] = lds16(&ldsK[32 + lg][db], &ldsK[32 + lg][db + 8]);
                kb0[3] = lds16(&ldsK[48 + lg][db], &ldsK[48 + lg][db + 8]);
            }
#pragma unroll
            for (int g = 0; g < 4; ++g) {
                v16h* cur = (g & 1) ? kb1 : kb0;
                v16h* nxt = (g & 1) ? kb0 : kb1;
                if (g < 3) {
                    // B 32x16 layout: lane n = lg -> kv row; elem e -> d = g*32 + hs*16 + e
                    const int db = (g + 1) * 32 + hs * 16;
                    nxt[0] = lds16(&ldsK[lg][db],      &ldsK[lg][db + 8]);
                    nxt[1] = lds16(&ldsK[16 + lg][db], &ldsK[16 + lg][db + 8]);
                    nxt[2] = lds16(&ldsK[32 + lg][db], &ldsK[32 + lg][db + 8]);
                    nxt[3] = lds16(&ldsK[48 + lg][db], &ldsK[48 + lg][db + 8]);
                }
                s[0] = __builtin_amdgcn_wmma_f32_16x16x32_f16(false, aq[g], false, cur[0],
                                                              (short)0, s[0], false, false);
                s[1] = __builtin_amdgcn_wmma_f32_16x16x32_f16(false, aq[g], false, cur[1],
                                                              (short)0, s[1], false, false);
                s[2] = __builtin_amdgcn_wmma_f32_16x16x32_f16(false, aq[g], false, cur[2],
                                                              (short)0, s[2], false, false);
                s[3] = __builtin_amdgcn_wmma_f32_16x16x32_f16(false, aq[g], false, cur[3],
                                                              (short)0, s[3], false, false);
            }

            // ---- scale + additive mask + causal ----
            // C layout: VGPR i -> q row (i + 8*hs); lane -> kv column t*16 + lg
            float pp[4][8];
#pragma unroll
            for (int i = 0; i < 8; ++i) {
                const int qr_  = qw + i + 8 * hs;
                const int qlim = PREF + qr_;
                const size_t mbase = (size_t)qr_ * KV + kvb + lg;
#pragma unroll
                for (int t = 0; t < 4; ++t) {
                    const int kvc = kvb + t * 16 + lg;
                    float x = s[t][i] * scale + mask[mbase + t * 16];
                    if (kvc > qlim) x = NEG_INF_F;
                    pp[t][i] = x;
                }
            }

            // ---- online softmax update (per q row) ----
#pragma unroll
            for (int i = 0; i < 8; ++i) {
                float mx = fmaxf(fmaxf(pp[0][i], pp[1][i]), fmaxf(pp[2][i], pp[3][i]));
                mx = redmax16(mx);
                float mn = fmaxf(mrow[i], mx);
                float al = __expf(mrow[i] - mn);  // -> 0 on first block
                float e0 = __expf(pp[0][i] - mn);
                float e1 = __expf(pp[1][i] - mn);
                float e2 = __expf(pp[2][i] - mn);
                float e3 = __expf(pp[3][i] - mn);
                float rs = redsum16((e0 + e1) + (e2 + e3));
                lrow[i] = lrow[i] * al + rs;
                mrow[i] = mn;
                pp[0][i] = e0;
                pp[1][i] = e1;
                pp[2][i] = e2;
                pp[3][i] = e3;
#pragma unroll
                for (int dc = 0; dc < 8; ++dc) o[dc][i] *= al;  // rescale O rows
            }

            // ---- P: C layout -> A layout via per-wave LDS scratch ----
#pragma unroll
            for (int i = 0; i < 8; ++i) {
                const int row = i + 8 * hs;
                h2 p01 = pkh2(pp[0][i], pp[1][i]);
                h2 p23 = pkh2(pp[2][i], pp[3][i]);
                ldsP[wave][row][lg]      = p01.x;
                ldsP[wave][row][16 + lg] = p01.y;
                ldsP[wave][row][32 + lg] = p23.x;
                ldsP[wave][row][48 + lg] = p23.y;
            }
            // same-wave LDS ops are in-order; compiler inserts DScnt waits
            v16h ap0 = lds16(&ldsP[wave][lg][hs * 16],      &ldsP[wave][lg][hs * 16 + 8]);
            v16h ap1 = lds16(&ldsP[wave][lg][32 + hs * 16], &ldsP[wave][lg][32 + hs * 16 + 8]);

            // ---- O += P V : 8 d-chunks x 2 kv-halves, double-buffered V tiles ----
            v16h vb0[2], vb1[2];
            {
                const int dr = lg;  // dc = 0
                vb0[0] = lds16(&ldsVt[dr][hs * 16],      &ldsVt[dr][hs * 16 + 8]);
                vb0[1] = lds16(&ldsVt[dr][32 + hs * 16], &ldsVt[dr][32 + hs * 16 + 8]);
            }
#pragma unroll
            for (int dc = 0; dc < 8; ++dc) {
                v16h* cur = (dc & 1) ? vb1 : vb0;
                v16h* nxt = (dc & 1) ? vb0 : vb1;
                if (dc < 7) {
                    const int dr = (dc + 1) * 16 + lg;  // B: lane n = d column in chunk
                    nxt[0] = lds16(&ldsVt[dr][hs * 16],      &ldsVt[dr][hs * 16 + 8]);
                    nxt[1] = lds16(&ldsVt[dr][32 + hs * 16], &ldsVt[dr][32 + hs * 16 + 8]);
                }
                o[dc] = __builtin_amdgcn_wmma_f32_16x16x32_f16(false, ap0, false, cur[0],
                                                               (short)0, o[dc], false, false);
                o[dc] = __builtin_amdgcn_wmma_f32_16x16x32_f16(false, ap1, false, cur[1],
                                                               (short)0, o[dc], false, false);
            }
        }
        __syncthreads();  // protect LDS tiles before next staging pass
    }

    // ---- epilogue: O / l, coalesced f32 stores ----
#pragma unroll
    for (int i = 0; i < 8; ++i) {
        const float inv = 1.0f / lrow[i];
        const int qr_ = qw + i + 8 * hs;
        float* op = out + ((size_t)h * Q + qr_) * D;
#pragma unroll
        for (int dc = 0; dc < 8; ++dc) op[dc * 16 + lg] = o[dc][i] * inv;
    }
}

extern "C" void kernel_launch(void* const* d_in, const int* in_sizes, int n_in,
                              void* d_out, int out_size, void* d_ws, size_t ws_size,
                              hipStream_t stream) {
    (void)in_sizes; (void)n_in; (void)out_size; (void)d_ws; (void)ws_size;
    const float* q    = (const float*)d_in[0];
    const float* k    = (const float*)d_in[1];
    const float* v    = (const float*)d_in[2];
    const float* mask = (const float*)d_in[3];
    float* out = (float*)d_out;

    dim3 grid(cfg::Q / cfg::QT, cfg::H, 1);  // (32, 16)
    dim3 block(256, 1, 1);                   // 8 waves (wave32)
    fsdpa_fwd_kernel<<<grid, block, 0, stream>>>(q, k, v, mask, out);
}
